// SelfKANtentionND_85323820302820
// MI455X (gfx1250) — matrix operand order repaired and checked by reading, not compiled, not run on hardware
//
#include <hip/hip_runtime.h>
#include <hip/hip_bf16.h>
#include <math.h>

// ---------------- types / helpers ----------------
typedef __attribute__((ext_vector_type(16))) __bf16 v16bf;
typedef __attribute__((ext_vector_type(8)))  float  v8f;

#define BATCH 4
#define CIN   128
#define CI    64
#define HH    64
#define WWID  64
#define NPIX  4096           // H*W
#define PSTR  (NPIX + 16)    // padded pixel stride; row NPIX is a zero ghost row
#define GSZ   8
#define CSPL  512            // CI*GSZ

union Frag { uint4 q[2]; unsigned u[8]; v16bf v; };

__device__ __forceinline__ unsigned short f2bf(float f) {
  union { float f; unsigned u; } c; c.f = f;
  unsigned u = c.u;
  return (unsigned short)((u + 0x7FFFu + ((u >> 16) & 1u)) >> 16);
}

// ---------------- stage 0: zero the ghost rows (harness poisons ws) ----------------
__global__ void k_zero_pad(unsigned short* __restrict__ siluT, unsigned short* __restrict__ basisT) {
  int idx = blockIdx.x * blockDim.x + threadIdx.x;   // BATCH*(CI+CSPL) = 2304
  if (idx < BATCH * CI) {
    int b = idx / CI, c = idx % CI;
    siluT[((size_t)b * PSTR + NPIX) * CI + c] = 0;
  } else {
    int r = idx - BATCH * CI;
    int b = r / CSPL, c = r % CSPL;
    basisT[((size_t)b * PSTR + NPIX) * CSPL + c] = 0;
  }
}

// ---------------- stage 1: 1x1 in-projection + silu (silu stored pixel-major, padded) ----------------
__global__ void k_proj_in(const float* __restrict__ x, const float* __restrict__ Win,
                          const float* __restrict__ bin, float* __restrict__ att,
                          unsigned short* __restrict__ siluT) {
  int idx = blockIdx.x * blockDim.x + threadIdx.x;     // over B*CI*NPIX
  int p  = idx & (NPIX - 1);
  int ci = (idx >> 12) & 63;
  int b  = idx >> 18;
  const float* xp = x + (size_t)b * CIN * NPIX + p;
  const float* wp = Win + ci * CIN;
  float acc = bin[ci];
#pragma unroll 4
  for (int c = 0; c < CIN; ++c) acc += wp[c] * xp[(size_t)c * NPIX];
  att[idx] = acc;
  siluT[((size_t)b * PSTR + p) * CI + ci] = f2bf(acc / (1.0f + __expf(-acc)));
}

// ---------------- stage 2: instance-norm stats per (b,c) ----------------
__global__ void k_inorm_stats(const float* __restrict__ att, float* __restrict__ st) {
  int bc = blockIdx.x;
  const float* p = att + (size_t)bc * NPIX;
  float s = 0.f, s2 = 0.f;
  for (int i = threadIdx.x; i < NPIX; i += 256) { float v = p[i]; s += v; s2 += v * v; }
  __shared__ float sh0[256], sh1[256];
  sh0[threadIdx.x] = s; sh1[threadIdx.x] = s2; __syncthreads();
  for (int o = 128; o > 0; o >>= 1) {
    if (threadIdx.x < o) { sh0[threadIdx.x] += sh0[threadIdx.x + o]; sh1[threadIdx.x] += sh1[threadIdx.x + o]; }
    __syncthreads();
  }
  if (threadIdx.x == 0) {
    float m = sh0[0] * (1.0f / NPIX);
    float v = sh1[0] * (1.0f / NPIX) - m * m;
    st[bc * 2] = m; st[bc * 2 + 1] = rsqrtf(v + 1e-5f);
  }
}

// ---------------- stage 3: RBF basis (pixel-major padded, one b128 store) ----------------
__global__ void k_basis(const float* __restrict__ att, const float* __restrict__ st,
                        unsigned short* __restrict__ basisT) {
  int idx = blockIdx.x * blockDim.x + threadIdx.x;     // over B*CI*NPIX
  int p = idx & 4095;
  int c = (idx >> 12) & 63;
  int b = idx >> 18;
  float m = st[(b * 64 + c) * 2], r = st[(b * 64 + c) * 2 + 1];
  float xn = (att[idx] - m) * r;
  const float invD = 7.0f / 4.0f;
  union { unsigned u[4]; uint4 q; } stv;
#pragma unroll
  for (int g2 = 0; g2 < 4; ++g2) {
    float g0 = -2.0f + (4.0f / 7.0f) * (float)(2 * g2);
    float g1 = -2.0f + (4.0f / 7.0f) * (float)(2 * g2 + 1);
    float t0 = (xn - g0) * invD, t1 = (xn - g1) * invD;
    stv.u[g2] = (unsigned)f2bf(__expf(-t0 * t0)) | ((unsigned)f2bf(__expf(-t1 * t1)) << 16);
  }
  *(uint4*)(basisT + ((size_t)b * PSTR + p) * CSPL + c * 8) = stv.q;
}

// ---------------- conv weight prepack into WMMA A-fragment layout ----------------
__global__ void k_packA(const float* __restrict__ W, unsigned int* __restrict__ Apack, int CinW) {
  int idx = blockIdx.x * blockDim.x + threadIdx.x;
  int j = idx & 7;
  int lane = (idx >> 3) & 31;
  int cotile = (idx >> 8) & 3;
  int chunk = idx >> 10;
  int ncb = CinW >> 5;
  int tap = chunk / ncb;
  int cb = chunk - tap * ncb;
  if (tap >= 9) return;
  int m = cotile * 16 + (lane & 15);
  int kloc = 2 * j + ((lane >= 16) ? 16 : 0);
  int ci0 = cb * 32 + kloc;
  size_t base = ((size_t)m * CinW) * 9 + tap;           // W is [co][ciW][3][3]
  float lo = W[base + (size_t)ci0 * 9];
  float hi = W[base + (size_t)(ci0 + 1) * 9];
  Apack[idx] = (unsigned)f2bf(lo) | ((unsigned)f2bf(hi) << 16);
}

// 1x1 W_out prepack: 2 K-chunks x 8 co-tiles
__global__ void k_packA1(const float* __restrict__ W, unsigned int* __restrict__ Apack) {
  int idx = blockIdx.x * blockDim.x + threadIdx.x;      // 2*8*32*8 = 4096
  int j = idx & 7;
  int lane = (idx >> 3) & 31;
  int cotile = (idx >> 8) & 7;
  int kk = idx >> 11;
  int m = cotile * 16 + (lane & 15);
  int kloc = ((j < 4) ? 2 * j : 16 + 2 * (j - 4)) + ((lane >= 16) ? 8 : 0);
  int ci = kk * 32 + kloc;
  float lo = W[m * CI + ci];
  float hi = W[m * CI + ci + 1];
  Apack[idx] = (unsigned)f2bf(lo) | ((unsigned)f2bf(hi) << 16);
}

// ---------------- stage 4: fused base+spline 3x3 conv (implicit GEMM, WMMA) ----------------
__global__ void __launch_bounds__(256) k_conv_fastkan(
    const unsigned short* __restrict__ siluT,   // [B][PSTR][64]
    const unsigned short* __restrict__ basisT,  // [B][PSTR][512]
    const unsigned int* __restrict__ ApackB,
    const unsigned int* __restrict__ ApackS,
    unsigned short* __restrict__ outbuf,
    int transposed)                             // 1: [B][NPIX][64], 0: [B][64][NPIX]
{
  int tid = threadIdx.x;
  int lane = tid & 31;
  int wv = tid >> 5;                 // 0..7
  int cotile = wv >> 1;              // 0..3
  int sub = wv & 1;                  // pixel sub-tile
  int blk = blockIdx.x;              // B * H * (W/32)
  int wseg = blk & 1;
  int h = (blk >> 1) & 63;
  int b = blk >> 7;
  int n = lane & 15;
  int w0 = wseg * 32 + sub * 16;
  int wpix = w0 + n;
  int kbase = (lane >= 16) ? 16 : 0;
  int hi8 = (lane >= 16) ? 8 : 0;

  v8f acc = {};
#pragma unroll
  for (int pass = 0; pass < 2; ++pass) {
    const unsigned short* inT = pass ? (basisT + (size_t)b * PSTR * CSPL)
                                     : (siluT + (size_t)b * PSTR * CI);
    const unsigned int* Ap = pass ? ApackS : ApackB;
    const int Cw = pass ? CSPL : CI;
    const int ncb = pass ? 16 : 2;
    int achunk = 0;
    for (int tap = 0; tap < 9; ++tap) {
      int dy = tap / 3 - 1, dx = tap % 3 - 1;
      int hh = h + dy, wn = wpix + dx;
      bool vld = (hh >= 0) & (hh < 64) & (wn >= 0) & (wn < 64);
      int poff = vld ? (hh * 64 + wn) : NPIX;          // ghost zero row
      const unsigned short* prow = inT + (size_t)poff * Cw + kbase;
      for (int cb = 0; cb < ncb; ++cb, ++achunk) {
        Frag Bf;
        const unsigned short* bp = prow + cb * 32;
        Bf.q[0] = *(const uint4*)bp;        // K = kbase+0..7
        Bf.q[1] = *(const uint4*)(bp + 8);  // K = kbase+8..15
        Frag Af;
        const unsigned int* ap = ApackB + 0; // placeholder to keep type
        ap = Ap + (((size_t)achunk * 4 + cotile) * 32 + lane) * 8;
        Af.q[0] = *(const uint4*)ap;
        Af.q[1] = *(const uint4*)(ap + 4);
        __builtin_prefetch(ap + 1024, 0, 1);   // next chunk's A fragments
        acc = __builtin_amdgcn_wmma_f32_16x16x32_bf16(false, Af.v, false, Bf.v,
                                                      (short)0, acc, false, false);
      }
    }
  }
  int p = h * 64 + w0 + n;
  if (transposed) {
    union { unsigned u[4]; uint4 q; } stv;
#pragma unroll
    for (int jj = 0; jj < 4; ++jj)
      stv.u[jj] = (unsigned)f2bf(acc[2 * jj]) | ((unsigned)f2bf(acc[2 * jj + 1]) << 16);
    *(uint4*)(outbuf + (size_t)(b * NPIX + p) * CI + cotile * 16 + hi8) = stv.q;
  } else {
    unsigned short* op = outbuf + (size_t)b * CI * NPIX +
                         (size_t)(cotile * 16 + hi8) * NPIX + p;
#pragma unroll
    for (int j = 0; j < 8; ++j) op[(size_t)j * NPIX] = f2bf(acc[j]);
  }
}

// ---------------- stage 5: fused flash attention (single pass, WMMA) ----------------
__global__ void __launch_bounds__(128) k_attn(
    const unsigned short* __restrict__ qT, const unsigned short* __restrict__ kT,
    const unsigned short* __restrict__ vb,   // [B][64][NPIX] channel-major
    unsigned short* __restrict__ attT)       // [B][NPIX][64] bf16 out
{
  __shared__ unsigned short plds[4][16][34];
  __shared__ float scl[4][16];
  int lane = threadIdx.x & 31;
  int wv = threadIdx.x >> 5;
  int tile = blockIdx.x * 4 + wv;
  int b = tile >> 8;
  int q0 = (tile & 255) * 16;
  int n = lane & 15;
  int kb16 = (lane >= 16) ? 16 : 0, hi8 = (lane >= 16) ? 8 : 0;

  const unsigned short* qrow = qT + ((size_t)b * NPIX + q0 + n) * CI;
  const unsigned short* kbase2 = kT + (size_t)b * NPIX * CI;
  const unsigned short* vp = vb + (size_t)b * CI * NPIX;

  Frag QA[2];
#pragma unroll
  for (int kk = 0; kk < 2; ++kk) {
    QA[kk].q[0] = *(const uint4*)(qrow + kk * 32 + hi8);        // K=0..7
    QA[kk].q[1] = *(const uint4*)(qrow + kk * 32 + hi8 + 16);   // K=16..23
  }

  float m[8], l[8];
#pragma unroll
  for (int j = 0; j < 8; ++j) { m[j] = -1e30f; l[j] = 0.f; }
  v8f oacc[4] = {};

  for (int key0 = 0; key0 < NPIX; key0 += 32) {
    v8f e0 = {}, e1 = {};
#pragma unroll
    for (int kc = 0; kc < 2; ++kc) {
      const unsigned short* krow = kbase2 + (size_t)(key0 + kc * 16 + n) * CI;
      v8f e = {};
#pragma unroll
      for (int kk = 0; kk < 2; ++kk) {
        Frag KB;
        KB.q[0] = *(const uint4*)(krow + kk * 32 + hi8);
        KB.q[1] = *(const uint4*)(krow + kk * 32 + hi8 + 16);
        e = __builtin_amdgcn_wmma_f32_16x16x32_bf16(false, QA[kk].v, false, KB.v,
                                                    (short)0, e, false, false);
      }
      if (kc == 0) e0 = e; else e1 = e;
    }
    // online softmax update per query row
#pragma unroll
    for (int j = 0; j < 8; ++j) {
      float loc = fmaxf(e0[j], e1[j]);
#pragma unroll
      for (int o = 1; o < 16; o <<= 1) loc = fmaxf(loc, __shfl_xor(loc, o, 32));
      float mn = fmaxf(m[j], loc);
      float f = __expf(m[j] - mn);
      float a0 = __expf(e0[j] - mn);
      float a1 = __expf(e1[j] - mn);
      float s = a0 + a1;
#pragma unroll
      for (int o = 1; o < 16; o <<= 1) s += __shfl_xor(s, o, 32);
      l[j] = l[j] * f + s;
      m[j] = mn;
      plds[wv][hi8 + j][n] = f2bf(a0);           // [query][key 0..15]
      plds[wv][hi8 + j][16 + n] = f2bf(a1);      // [query][key 16..31]
      if (n == 0) scl[wv][hi8 + j] = f;
    }
    asm volatile("s_wait_dscnt 0" ::: "memory");
    float fcol = scl[wv][n];                     // per-query rescale of accumulators
#pragma unroll
    for (int ct = 0; ct < 4; ++ct)
#pragma unroll
      for (int j = 0; j < 8; ++j) oacc[ct][j] *= fcol;

    Frag PB;                                     // B: K=keys, N=queries
#pragma unroll
    for (int j = 0; j < 8; ++j) {
      int k = ((j < 4) ? 2 * j : 16 + 2 * (j - 4)) + hi8;
      PB.u[j] = *(const unsigned int*)&plds[wv][n][k];
    }
#pragma unroll
    for (int ct = 0; ct < 4; ++ct) {
      Frag VA;                                   // A: M=channels, K=keys
      const unsigned short* vr = vp + (size_t)(ct * 16 + n) * NPIX + key0;
#pragma unroll
      for (int j = 0; j < 8; ++j) {
        int k = ((j < 4) ? 2 * j : 16 + 2 * (j - 4)) + hi8;
        VA.u[j] = *(const unsigned int*)(vr + k);
      }
      oacc[ct] = __builtin_amdgcn_wmma_f32_16x16x32_bf16(false, VA.v, false, PB.v,
                                                         (short)0, oacc[ct], false, false);
    }
  }
  // final 1/l normalization (per query = per lane) and bf16 pixel-major store
  if (n == 0) {
#pragma unroll
    for (int j = 0; j < 8; ++j) scl[wv][hi8 + j] = 1.0f / l[j];
  }
  asm volatile("s_wait_dscnt 0" ::: "memory");
  float rl = scl[wv][n];
  unsigned short* orow = attT + ((size_t)b * NPIX + q0 + n) * CI;
#pragma unroll
  for (int ct = 0; ct < 4; ++ct) {
    union { unsigned u[4]; uint4 q; } stv;
#pragma unroll
    for (int jj = 0; jj < 4; ++jj)
      stv.u[jj] = (unsigned)f2bf(oacc[ct][2 * jj] * rl) |
                  ((unsigned)f2bf(oacc[ct][2 * jj + 1] * rl) << 16);
    *(uint4*)(orow + ct * 16 + hi8) = stv.q;
  }
}

// ---------------- stage 6: 1x1 out-projection + residual (WMMA) ----------------
__global__ void __launch_bounds__(256) k_proj_out_wmma(
    const unsigned short* __restrict__ attT,   // [B][NPIX][64] bf16
    const unsigned int* __restrict__ ApO,      // packed W_out fragments
    const float* __restrict__ bout, const float* __restrict__ x,
    const float* __restrict__ gamma, float* __restrict__ ybufT) {  // [B][NPIX][128] f32
  int lane = threadIdx.x & 31;
  int wv = threadIdx.x >> 5;          // co-tile 0..7 (M=128)
  int blk = blockIdx.x;               // B * NPIX/32
  int seg = blk & 127;
  int b = blk >> 7;
  int p0 = seg * 32;
  int n = lane & 15;
  int kb16 = (lane >= 16) ? 16 : 0, hi8 = (lane >= 16) ? 8 : 0;
  const unsigned short* at = attT + (size_t)b * NPIX * CI;
  float g = gamma[0];

  v8f acc[2] = {};
#pragma unroll
  for (int kk = 0; kk < 2; ++kk) {
    Frag Af;
    const unsigned int* ap = ApO + (((size_t)kk * 8 + wv) * 32 + lane) * 8;
    Af.q[0] = *(const uint4*)ap;
    Af.q[1] = *(const uint4*)(ap + 4);
#pragma unroll
    for (int sub = 0; sub < 2; ++sub) {
      const unsigned short* arow = at + (size_t)(p0 + sub * 16 + n) * CI;
      Frag Bf;
      Bf.q[0] = *(const uint4*)(arow + kk * 32 + hi8);
      Bf.q[1] = *(const uint4*)(arow + kk * 32 + hi8 + 16);
      acc[sub] = __builtin_amdgcn_wmma_f32_16x16x32_bf16(false, Af.v, false, Bf.v,
                                                         (short)0, acc[sub], false, false);
    }
  }
  int co0 = wv * 16 + hi8;
#pragma unroll
  for (int sub = 0; sub < 2; ++sub) {
    int pix = p0 + sub * 16 + n;
    const float* xcol = x + (size_t)b * CIN * NPIX + (size_t)co0 * NPIX + pix;
    float* yrow = ybufT + ((size_t)b * NPIX + pix) * CIN + co0;
    float v[8];
#pragma unroll
    for (int j = 0; j < 8; ++j)
      v[j] = acc[sub][j] + bout[co0 + j] + g * xcol[(size_t)j * NPIX];
    *(float4*)yrow = make_float4(v[0], v[1], v[2], v[3]);
    *(float4*)(yrow + 4) = make_float4(v[4], v[5], v[6], v[7]);
  }
}

// ---------------- stage 7: batchnorm (ybuf pixel-major) ----------------
__global__ void k_bn_stats(const float* __restrict__ ybufT, float* __restrict__ st) {
  int c = blockIdx.x;
  float s = 0.f, s2 = 0.f;
  for (int i = threadIdx.x; i < BATCH * NPIX; i += 256) {
    float v = ybufT[(size_t)i * CIN + c];
    s += v; s2 += v * v;
  }
  __shared__ float sh0[256], sh1[256];
  sh0[threadIdx.x] = s; sh1[threadIdx.x] = s2; __syncthreads();
  for (int o = 128; o > 0; o >>= 1) {
    if (threadIdx.x < o) { sh0[threadIdx.x] += sh0[threadIdx.x + o]; sh1[threadIdx.x] += sh1[threadIdx.x + o]; }
    __syncthreads();
  }
  if (threadIdx.x == 0) {
    float m = sh0[0] * (1.0f / (BATCH * NPIX));
    float v = sh1[0] * (1.0f / (BATCH * NPIX)) - m * m;
    st[c * 2] = m; st[c * 2 + 1] = rsqrtf(v + 1e-5f);
  }
}

__global__ void k_bn_apply(const float* __restrict__ ybufT, const float* __restrict__ st,
                           const float* __restrict__ bnw, const float* __restrict__ bnb,
                           float* __restrict__ out) {
  int idx = blockIdx.x * blockDim.x + threadIdx.x;   // NCHW output order
  int p = idx & 4095;
  int c = (idx >> 12) & 127;
  int b = idx >> 19;
  float v = ybufT[((size_t)b * NPIX + p) * CIN + c];
  out[idx] = (v - st[c * 2]) * st[c * 2 + 1] * bnw[c] + bnb[c];
}

// ---------------- launch ----------------
extern "C" void kernel_launch(void* const* d_in, const int* in_sizes, int n_in,
                              void* d_out, int out_size, void* d_ws, size_t ws_size,
                              hipStream_t stream) {
  const float* x       = (const float*)d_in[0];
  const float* W_in    = (const float*)d_in[1];
  const float* b_in    = (const float*)d_in[2];
  const float* W_out   = (const float*)d_in[3];
  const float* b_out   = (const float*)d_in[4];
  const float* Wq_b    = (const float*)d_in[5];
  const float* Wq_s    = (const float*)d_in[6];
  const float* Wk_b    = (const float*)d_in[7];
  const float* Wk_s    = (const float*)d_in[8];
  const float* Wv_b    = (const float*)d_in[9];
  const float* Wv_s    = (const float*)d_in[10];
  const float* gamma   = (const float*)d_in[11];
  const float* bn_w    = (const float*)d_in[12];
  const float* bn_b    = (const float*)d_in[13];

  char* ws = (char*)d_ws;
  size_t o = 0;
  auto alloc = [&](size_t bytes) { size_t r = o; o += (bytes + 255) & ~(size_t)255; return r; };

  float*          att_in = (float*)(ws + alloc((size_t)BATCH * CI * NPIX * 4));
  unsigned short* siluT  = (unsigned short*)(ws + alloc((size_t)BATCH * PSTR * CI * 2));
  float*          instSt = (float*)(ws + alloc((size_t)BATCH * CI * 2 * 4));
  unsigned short* basisT = (unsigned short*)(ws + alloc((size_t)BATCH * PSTR * CSPL * 2));
  unsigned short* qbufT  = (unsigned short*)(ws + alloc((size_t)BATCH * NPIX * CI * 2));
  unsigned short* kbufT  = (unsigned short*)(ws + alloc((size_t)BATCH * NPIX * CI * 2));
  unsigned short* vbuf   = (unsigned short*)(ws + alloc((size_t)BATCH * CI * NPIX * 2));
  unsigned short* attT   = (unsigned short*)(ws + alloc((size_t)BATCH * NPIX * CI * 2));
  float*          ybufT  = (float*)(ws + alloc((size_t)BATCH * NPIX * CIN * 4));
  float*          bnSt   = (float*)(ws + alloc((size_t)CIN * 2 * 4));
  const size_t baseEnt   = (size_t)9 * 2 * 4 * 32 * 8;     // 18432 u32
  const size_t splEnt    = (size_t)9 * 16 * 4 * 32 * 8;    // 147456 u32
  const size_t outEnt    = (size_t)2 * 8 * 32 * 8;         // 4096 u32
  unsigned int* ApQb = (unsigned int*)(ws + alloc(baseEnt * 4));
  unsigned int* ApQs = (unsigned int*)(ws + alloc(splEnt * 4));
  unsigned int* ApKb = (unsigned int*)(ws + alloc(baseEnt * 4));
  unsigned int* ApKs = (unsigned int*)(ws + alloc(splEnt * 4));
  unsigned int* ApVb = (unsigned int*)(ws + alloc(baseEnt * 4));
  unsigned int* ApVs = (unsigned int*)(ws + alloc(splEnt * 4));
  unsigned int* ApO  = (unsigned int*)(ws + alloc(outEnt * 4));

  // stage 0-3
  k_zero_pad<<<(BATCH * (CI + CSPL)) / 256, 256, 0, stream>>>(siluT, basisT);
  k_proj_in<<<(BATCH * CI * NPIX) / 256, 256, 0, stream>>>(x, W_in, b_in, att_in, siluT);
  k_inorm_stats<<<BATCH * CI, 256, 0, stream>>>(att_in, instSt);
  k_basis<<<(BATCH * CI * NPIX) / 256, 256, 0, stream>>>(att_in, instSt, basisT);

  // weight prepack
  k_packA<<<(int)(baseEnt / 256), 256, 0, stream>>>(Wq_b, ApQb, CI);
  k_packA<<<(int)(splEnt / 256), 256, 0, stream>>>(Wq_s, ApQs, CSPL);
  k_packA<<<(int)(baseEnt / 256), 256, 0, stream>>>(Wk_b, ApKb, CI);
  k_packA<<<(int)(splEnt / 256), 256, 0, stream>>>(Wk_s, ApKs, CSPL);
  k_packA<<<(int)(baseEnt / 256), 256, 0, stream>>>(Wv_b, ApVb, CI);
  k_packA<<<(int)(splEnt / 256), 256, 0, stream>>>(Wv_s, ApVs, CSPL);
  k_packA1<<<(int)(outEnt / 256), 256, 0, stream>>>(W_out, ApO);

  // stage 4: q, k, v fastkan convs (WMMA implicit GEMM)
  const int convBlocks = BATCH * HH * (WWID / 32);   // 512
  k_conv_fastkan<<<convBlocks, 256, 0, stream>>>(siluT, basisT, ApQb, ApQs, qbufT, 1);
  k_conv_fastkan<<<convBlocks, 256, 0, stream>>>(siluT, basisT, ApKb, ApKs, kbufT, 1);
  k_conv_fastkan<<<convBlocks, 256, 0, stream>>>(siluT, basisT, ApVb, ApVs, vbuf, 0);

  // stage 5: fused flash attention (WMMA)
  k_attn<<<(BATCH * (NPIX / 16)) / 4, 128, 0, stream>>>(qbufT, kbufT, vbuf, attT);

  // stage 6: 1x1 out-projection + residual (WMMA)
  k_proj_out_wmma<<<BATCH * (NPIX / 32), 256, 0, stream>>>(attT, ApO, b_out, x, gamma, ybufT);

  // stage 7: batchnorm
  k_bn_stats<<<CIN, 256, 0, stream>>>(ybufT, bnSt);
  k_bn_apply<<<(BATCH * CIN * NPIX) / 256, 256, 0, stream>>>(ybufT, bnSt, bn_w, bn_b, (float*)d_out);
}
